// ProtoContrastLoss_11175504904813
// MI455X (gfx1250) — matrix operand order repaired and checked by reading, not compiled, not run on hardware
//
#include <hip/hip_runtime.h>
#include <math.h>

// Problem constants (from reference): B=1024, K=16, D=256, N=2B=2048
#define BB 1024
#define NN 2048
#define KK 16
#define DD 256
#define MAX_TILES 144          // <= 2048/16 + 15 padding tiles
#define PERM_PAD (MAX_TILES*16)

typedef __attribute__((ext_vector_type(16))) __bf16        v16bf;
typedef __attribute__((ext_vector_type(8)))  float         v8f;
typedef __attribute__((ext_vector_type(8)))  unsigned int  v8u;

__device__ __forceinline__ unsigned short f2bf(float f) {
  unsigned int u = __float_as_uint(f);
  unsigned int r = u + 0x7FFFu + ((u >> 16) & 1u);   // round-to-nearest-even
  return (unsigned short)(r >> 16);
}

// ---------------- small prep kernels ----------------

__global__ void k_init(int* counts, int* fills, int* perm) {
  int t = blockIdx.x * blockDim.x + threadIdx.x;
  if (t < KK) { counts[t] = 0; fills[t] = 0; }
  if (t < PERM_PAD) perm[t] = -1;
}

__global__ void k_argmax(const float* __restrict__ w, const float* __restrict__ g,
                         int* __restrict__ idx, int* counts) {
  int b = blockIdx.x * blockDim.x + threadIdx.x;
  if (b >= BB) return;
  float best = -INFINITY; int bi = 0;
  for (int k = 0; k < KK; ++k) {
    float v = (logf(w[b * KK + k] + 1e-12f) + g[b * KK + k]) * 1.25f;  // / tau=0.8
    if (v > best) { best = v; bi = k; }
  }
  idx[b] = bi;
  atomicAdd(&counts[bi], 2);   // anchor b and b+B share the cluster
}

__global__ void k_pnorm(const float* __restrict__ prototypes, float* __restrict__ protos) {
  __shared__ float red[256];
  int k = blockIdx.x, t = threadIdx.x;
  float v = prototypes[k * DD + t];
  red[t] = v * v; __syncthreads();
  for (int s = 128; s > 0; s >>= 1) { if (t < s) red[t] += red[t + s]; __syncthreads(); }
  float norm = fmaxf(sqrtf(red[0]), 1e-12f);
  protos[k * DD + t] = v / norm;
}

__global__ void k_scan(const int* __restrict__ counts, int* padoff, int* tilecl, int* ptiles) {
  if (threadIdx.x | blockIdx.x) return;
  int off = 0;
  for (int k = 0; k < KK; ++k) {
    padoff[k] = off;
    int nt = (counts[k] + 15) >> 4;
    for (int i = 0; i < nt; ++i) tilecl[(off >> 4) + i] = k;
    off += nt << 4;
  }
  ptiles[0] = off >> 4;
}

__global__ void k_scatter(const int* __restrict__ idx, const int* __restrict__ padoff,
                          int* fills, int* perm) {
  int a = blockIdx.x * blockDim.x + threadIdx.x;
  if (a >= NN) return;
  int k = idx[a & (BB - 1)];
  int p = atomicAdd(&fills[k], 1);
  perm[padoff[k] + p] = a;
}

__global__ void k_zcat(const float* __restrict__ p1, const float* __restrict__ p2,
                       unsigned short* __restrict__ zcat) {
  int t = blockIdx.x * blockDim.x + threadIdx.x;   // over NN*DD
  int r = t >> 8, c = t & 255;
  float v = (r < BB) ? p1[r * DD + c] : p2[(r - BB) * DD + c];
  zcat[t] = f2bf(v);
}

__global__ void k_aperm(const int* __restrict__ perm, const int* __restrict__ idx,
                        const float* __restrict__ zi, const float* __restrict__ zj,
                        unsigned short* __restrict__ ap) {
  int t = blockIdx.x * blockDim.x + threadIdx.x;   // over PERM_PAD*DD
  int s = t >> 8, c = t & 255;
  int a = perm[s];
  float v = 0.f;
  if (a >= 0) {
    int b = a & (BB - 1);
    int k = idx[b];
    const float* src = (a < BB ? zi : zj) + ((size_t)(b * KK + k)) * DD;
    v = src[c];
  }
  ap[t] = f2bf(v);
}

__global__ void k_pos(const int* __restrict__ idx, const float* __restrict__ zi,
                      const float* __restrict__ zj, const float* __restrict__ protos,
                      float* __restrict__ posp, float* __restrict__ posv) {
  int wid  = (blockIdx.x * blockDim.x + threadIdx.x) >> 5;  // one wave32 per anchor
  int lane = threadIdx.x & 31;
  if (wid >= NN) return;
  int b = wid & (BB - 1);
  int k = idx[b];
  size_t ro = ((size_t)(b * KK + k)) * DD;
  const float* za = (wid < BB ? zi : zj) + ro;
  const float* zc = (wid < BB ? zj : zi) + ro;
  const float* pr = protos + k * DD;
  float dp = 0.f, dv = 0.f;
#pragma unroll
  for (int j = 0; j < 8; ++j) {
    int d = lane + 32 * j;
    float av = za[d];
    dp += av * pr[d]; dv += av * zc[d];
  }
#pragma unroll
  for (int off = 16; off; off >>= 1) {
    dp += __shfl_xor(dp, off, 32);
    dv += __shfl_xor(dv, off, 32);
  }
  if (!lane) { posp[wid] = dp * 5.0f; posv[wid] = dv * 5.0f; }   // / T_PROTO=0.2
}

// ---------------- WMMA GEMMs ----------------

// sim = zcat @ zcatT ; 8 waves/block; block tile 32 rows x 64 cols
__global__ void k_simgemm(const unsigned short* __restrict__ zcat, float* __restrict__ simmat) {
  int w = threadIdx.x >> 5, lane = threadIdx.x & 31;
  int ra = blockIdx.x * 32 + (w & 1) * 16;
  int cb = blockIdx.y * 64 + (w >> 1) * 16;
  int mr = lane & 15, h = lane >> 4;
  const unsigned short* Arow = zcat + (size_t)(ra + mr) * DD;
  const unsigned short* Brow = zcat + (size_t)(cb + mr) * DD;
  v8f c = {0.f, 0.f, 0.f, 0.f, 0.f, 0.f, 0.f, 0.f};
  for (int kb = 0; kb < DD; kb += 32) {
    v8u araw, braw;
#pragma unroll
    for (int p = 0; p < 8; ++p) {
      int koff = ((p < 4) ? (p * 2) : (16 + (p - 4) * 2)) + h * 8;   // A interleave
      araw[p] = *(const unsigned int*)(Arow + kb + koff);
      braw[p] = *(const unsigned int*)(Brow + kb + koff);  // B interleave (same as A)
    }
    v16bf af = __builtin_bit_cast(v16bf, araw);
    v16bf bfr = __builtin_bit_cast(v16bf, braw);
    c = __builtin_amdgcn_wmma_f32_16x16x32_bf16(false, af, false, bfr, (short)0, c, false, false);
  }
#pragma unroll
  for (int p = 0; p < 8; ++p) {
    int m = p + 8 * h;
    simmat[(size_t)(ra + m) * NN + cb + mr] = c[p];
  }
}

// neg[a, :] : grouped-by-cluster anchors vs z_i / z_j rows at the tile's cluster
__global__ void k_neggemm(const unsigned short* __restrict__ aperm, const int* __restrict__ perm,
                          const int* __restrict__ tilecl, const int* __restrict__ ptiles,
                          const float* __restrict__ zi, const float* __restrict__ zj,
                          float* __restrict__ negmat) {
  int stile = blockIdx.x;
  if (stile >= ptiles[0]) return;
  int k = tilecl[stile];
  int w = threadIdx.x >> 5, lane = threadIdx.x & 31;
  int ctile = blockIdx.y * 4 + w;                 // 0..127 : [neg_i | neg_j]
  const float* zsrc = (ctile < 64) ? zi : zj;
  int b0 = (ctile & 63) * 16;
  int n = lane & 15, h = lane >> 4;
  const unsigned short* Arow = aperm + (size_t)(stile * 16 + n) * DD;
  const float* Brow = zsrc + ((size_t)((b0 + n) * KK + k)) * DD;
  v8f c = {0.f, 0.f, 0.f, 0.f, 0.f, 0.f, 0.f, 0.f};
  for (int kb = 0; kb < DD; kb += 32) {
    v8u araw, braw;
#pragma unroll
    for (int p = 0; p < 8; ++p) {
      int koff = ((p < 4) ? (p * 2) : (16 + (p - 4) * 2)) + h * 8;
      araw[p] = *(const unsigned int*)(Arow + kb + koff);
      int kk = kb + koff;
      braw[p] = (unsigned int)f2bf(Brow[kk]) | ((unsigned int)f2bf(Brow[kk + 1]) << 16);
    }
    v16bf af = __builtin_bit_cast(v16bf, araw);
    v16bf bfr = __builtin_bit_cast(v16bf, braw);
    c = __builtin_amdgcn_wmma_f32_16x16x32_bf16(false, af, false, bfr, (short)0, c, false, false);
  }
#pragma unroll
  for (int p = 0; p < 8; ++p) {
    int m = p + 8 * h;
    int a = perm[stile * 16 + m];
    if (a >= 0) negmat[(size_t)a * NN + ctile * 16 + n] = c[p];
  }
}

// ---------------- reductions ----------------

__global__ void k_ntred(const float* __restrict__ simmat, float* __restrict__ pnt) {
  __shared__ float red[256];
  int a = blockIdx.x, t = threadIdx.x;
  int partner = (a + BB) & (NN - 1);
  const float* row = simmat + (size_t)a * NN;
  float lmax = -INFINITY;
  for (int j = t; j < NN; j += 256)
    if (j != a && j != partner) lmax = fmaxf(lmax, row[j] * 2.0f);     // / T_SIM=0.5
  red[t] = lmax; __syncthreads();
  for (int s = 128; s; s >>= 1) { if (t < s) red[t] = fmaxf(red[t], red[t + s]); __syncthreads(); }
  float M = red[0]; __syncthreads();
  float ls = 0.f;
  for (int j = t; j < NN; j += 256)
    if (j != a && j != partner) ls += expf(row[j] * 2.0f - M);
  red[t] = ls; __syncthreads();
  for (int s = 128; s; s >>= 1) { if (t < s) red[t] += red[t + s]; __syncthreads(); }
  if (!t) {
    float lse_neg = M + logf(red[0]);
    float pos = row[partner] * 2.0f;
    float mm = fmaxf(pos, lse_neg);
    float lse_all = mm + logf(expf(pos - mm) + expf(lse_neg - mm));
    pnt[a] = lse_all - pos;
  }
}

__global__ void k_prred(const float* __restrict__ negmat, const int* __restrict__ idx,
                        const float* __restrict__ posp, const float* __restrict__ posv,
                        float* __restrict__ ppr) {
  __shared__ float red[256];
  int a = blockIdx.x, t = threadIdx.x;
  int k = idx[a & (BB - 1)];
  float pp = posp[a], pv = posv[a];
  const float* row = negmat + (size_t)a * NN;
  float lmax = -INFINITY;
  for (int j = t; j < NN; j += 256)
    if (idx[j & (BB - 1)] != k) lmax = fmaxf(lmax, row[j] * 5.0f);     // / T_PROTO=0.2
  red[t] = lmax; __syncthreads();
  for (int s = 128; s; s >>= 1) { if (t < s) red[t] = fmaxf(red[t], red[t + s]); __syncthreads(); }
  float M = fmaxf(fmaxf(red[0], pp), pv); __syncthreads();
  float ls = 0.f;
  for (int j = t; j < NN; j += 256)
    if (idx[j & (BB - 1)] != k) ls += expf(row[j] * 5.0f - M);
  red[t] = ls; __syncthreads();
  for (int s = 128; s; s >>= 1) { if (t < s) red[t] += red[t + s]; __syncthreads(); }
  if (!t) {
    float total = red[0] + expf(pp - M) + expf(pv - M);
    float lse = M + logf(total);
    float m2 = fmaxf(pp, pv);
    ppr[a] = lse - (m2 + logf(expf(pp - m2) + expf(pv - m2)));
  }
}

__global__ void k_final(const float* __restrict__ pnt, const float* __restrict__ ppr,
                        float* __restrict__ out) {
  __shared__ float red[256];
  int t = threadIdx.x;
  float s1 = 0.f, s2 = 0.f;
  for (int j = t; j < NN; j += 256) { s1 += pnt[j]; s2 += ppr[j]; }
  red[t] = s1; __syncthreads();
  for (int s = 128; s; s >>= 1) { if (t < s) red[t] += red[t + s]; __syncthreads(); }
  float S1 = red[0]; __syncthreads();
  red[t] = s2; __syncthreads();
  for (int s = 128; s; s >>= 1) { if (t < s) red[t] += red[t + s]; __syncthreads(); }
  if (!t) out[0] = S1 / (float)NN + red[0] / (float)NN;
}

// ---------------- launch ----------------

extern "C" void kernel_launch(void* const* d_in, const int* in_sizes, int n_in,
                              void* d_out, int out_size, void* d_ws, size_t ws_size,
                              hipStream_t stream) {
  const float* proj1 = (const float*)d_in[0];
  const float* proj2 = (const float*)d_in[1];
  const float* z_i   = (const float*)d_in[2];
  const float* z_j   = (const float*)d_in[3];
  const float* w_i   = (const float*)d_in[4];
  const float* proto = (const float*)d_in[5];
  const float* gumb  = (const float*)d_in[6];
  float* out = (float*)d_out;

  unsigned char* ws = (unsigned char*)d_ws;
  size_t off = 0;
  auto take = [&](size_t bytes) -> void* {
    void* p = ws + off;
    off += (bytes + 255) & ~(size_t)255;
    return p;
  };
  int*   idx    = (int*)  take((size_t)BB * 4);
  int*   counts = (int*)  take(KK * 4);
  int*   fills  = (int*)  take(KK * 4);
  int*   padoff = (int*)  take(KK * 4);
  int*   tilecl = (int*)  take(MAX_TILES * 4);
  int*   ptiles = (int*)  take(4);
  int*   perm   = (int*)  take((size_t)PERM_PAD * 4);
  float* protos = (float*)take((size_t)KK * DD * 4);
  float* posp   = (float*)take((size_t)NN * 4);
  float* posv   = (float*)take((size_t)NN * 4);
  unsigned short* zcat  = (unsigned short*)take((size_t)NN * DD * 2);
  unsigned short* aperm = (unsigned short*)take((size_t)PERM_PAD * DD * 2);
  float* simmat = (float*)take((size_t)NN * NN * 4);
  float* negmat = (float*)take((size_t)NN * NN * 4);
  float* pnt    = (float*)take((size_t)NN * 4);
  float* ppr    = (float*)take((size_t)NN * 4);
  (void)ws_size; (void)in_sizes; (void)n_in; (void)out_size;

  k_init   <<<dim3((PERM_PAD + 255) / 256), 256, 0, stream>>>(counts, fills, perm);
  k_argmax <<<dim3(BB / 256), 256, 0, stream>>>(w_i, gumb, idx, counts);
  k_pnorm  <<<dim3(KK), 256, 0, stream>>>(proto, protos);
  k_scan   <<<dim3(1), 32, 0, stream>>>(counts, padoff, tilecl, ptiles);
  k_scatter<<<dim3(NN / 256), 256, 0, stream>>>(idx, padoff, fills, perm);
  k_zcat   <<<dim3(NN * DD / 256), 256, 0, stream>>>(proj1, proj2, zcat);
  k_aperm  <<<dim3(PERM_PAD * DD / 256), 256, 0, stream>>>(perm, idx, z_i, z_j, aperm);
  k_pos    <<<dim3(NN / 8), 256, 0, stream>>>(idx, z_i, z_j, protos, posp, posv);
  k_simgemm<<<dim3(NN / 32, NN / 64), 256, 0, stream>>>(zcat, simmat);
  k_neggemm<<<dim3(MAX_TILES, NN / 64), 128, 0, stream>>>(aperm, perm, tilecl, ptiles, z_i, z_j, negmat);
  k_ntred  <<<dim3(NN), 256, 0, stream>>>(simmat, pnt);
  k_prred  <<<dim3(NN), 256, 0, stream>>>(negmat, idx, posp, posv, ppr);
  k_final  <<<dim3(1), 256, 0, stream>>>(pnt, ppr, out);
}
